// MyRNN_78168404787652
// MI455X (gfx1250) — compile-verified, matrix operand order checked by reading, not run on hardware
//
#include <hip/hip_runtime.h>
#include <hip/hip_bf16.h>

// ---------------------------------------------------------------------------
// RNN (tanh) for MI455X / gfx1250, wave32 + WMMA bf16.
//   BATCH=64, SEQ=512, INPUT=1024, HIDDEN=1024
// Phase 0 (prep):   transpose W -> WxT/WhT (bf16, [N][K]), h0 -> bf16, bar=0
// Phase 1 (gemm):   x_proj = input @ Wx + b, software-pipelined WMMA bf16,
//                   hardware f32->bf16 conversion fused into A-frag loads
// Phase 2 (recur):  persistent kernel; WhT tiles LDS-resident for all 512
//                   steps, h staged global->LDS per step via
//                   global_load_async_to_lds_b128 (ASYNCcnt), WMMA chain
//                   reads LDS only; device-scope grid barrier per step.
// ---------------------------------------------------------------------------

#define BATCH   64
#define SEQ     512
#define INSZ    1024
#define HID     1024
#define NBLK_REC 32        // persistent blocks (32 blk * 8 waves = 256 tiles)

#define SMEM_WH_ELEMS (32 * HID)        // 32 WhT rows per block  (64 KB)
#define SMEM_H_ELEMS  (BATCH * HID)     // full hidden state      (128 KB)
#define SMEM_BYTES    ((SMEM_WH_ELEMS + SMEM_H_ELEMS) * 2)   // 192 KB < 320 KB

typedef __bf16 bf16_t;
typedef __attribute__((ext_vector_type(16))) __bf16 v16bf;
typedef __attribute__((ext_vector_type(8)))  __bf16 v8bf;
typedef __attribute__((ext_vector_type(8)))  float  v8f;
typedef __attribute__((ext_vector_type(4)))  float  v4f;
typedef __attribute__((ext_vector_type(4)))  int    v4i;

typedef __attribute__((address_space(1))) v4i* gptr_v4i;   // global int4*
typedef __attribute__((address_space(3))) v4i* lptr_v4i;   // LDS int4*

// ---- fp32 -> bf16: let the backend use the hardware converter (RNE) ------
__device__ __forceinline__ bf16_t to_bf16(float f) { return (bf16_t)f; }

// ---- global -> LDS staging (async-to-LDS path when available) -------------
#if __has_builtin(__builtin_amdgcn_global_load_async_to_lds_b128)
#define HAS_ASYNC_LDS 1
#else
#define HAS_ASYNC_LDS 0
#endif

__device__ __forceinline__ void stage_16B(const bf16_t* gsrc, bf16_t* ldst) {
#if HAS_ASYNC_LDS
    __builtin_amdgcn_global_load_async_to_lds_b128(
        (gptr_v4i)(void*)const_cast<bf16_t*>(gsrc),
        (lptr_v4i)(void*)ldst, 0, 0);
#else
    *(uint4*)ldst = *(const uint4*)gsrc;
#endif
}

__device__ __forceinline__ void stage_wait() {
#if HAS_ASYNC_LDS
#if __has_builtin(__builtin_amdgcn_s_wait_asynccnt)
    __builtin_amdgcn_s_wait_asynccnt(0);
#else
    asm volatile("s_wait_asynccnt 0x0" ::: "memory");
#endif
#endif
}

// ---- fragment loads -------------------------------------------------------
// WMMA 16-bit operand layout (wave32): lane l holds row/col (l&15), with
// K elements [kb..kb+7] in vector elems 0..7 and [kb+16..kb+23] in 8..15,
// kb = k0 + 8*(l>>4).  Both 8-element chunks are 16B-aligned (b128).
__device__ __forceinline__ v16bf load_frag_bf16(const bf16_t* rowbase, int kb) {
    v8bf lo = *(const v8bf*)(rowbase + kb);
    v8bf hi = *(const v8bf*)(rowbase + kb + 16);
    v16bf r;
#pragma unroll
    for (int i = 0; i < 8; ++i) { r[i] = lo[i]; r[i + 8] = hi[i]; }
    return r;
}

// Same layout but sourced from fp32 (hardware cvt fused into the load).
__device__ __forceinline__ v16bf load_frag_f32(const float* rowbase, int kb) {
    v4f a0 = *(const v4f*)(rowbase + kb);
    v4f a1 = *(const v4f*)(rowbase + kb + 4);
    v4f b0 = *(const v4f*)(rowbase + kb + 16);
    v4f b1 = *(const v4f*)(rowbase + kb + 20);
    v16bf r;
#pragma unroll
    for (int i = 0; i < 4; ++i) {
        r[i]      = (bf16_t)a0[i];
        r[i + 4]  = (bf16_t)a1[i];
        r[i + 8]  = (bf16_t)b0[i];
        r[i + 12] = (bf16_t)b1[i];
    }
    return r;
}

__device__ __forceinline__ v8f wmma_bf16(v16bf a, v16bf b, v8f c) {
    return __builtin_amdgcn_wmma_f32_16x16x32_bf16(
        /*neg_a=*/false, a, /*neg_b=*/false, b,
        /*c_mod=*/(short)0, c, /*reuse_a=*/false, /*reuse_b=*/false);
}

// ---- device-scope grid barrier (generation counting) ----------------------
__device__ __forceinline__ void grid_barrier(unsigned* bar, unsigned gen_target) {
    __syncthreads();
    if (threadIdx.x == 0) {
        unsigned prev = __hip_atomic_fetch_add(&bar[0], 1u, __ATOMIC_ACQ_REL,
                                               __HIP_MEMORY_SCOPE_AGENT);
        if (prev == (unsigned)(NBLK_REC - 1)) {
            __hip_atomic_store(&bar[0], 0u, __ATOMIC_RELAXED, __HIP_MEMORY_SCOPE_AGENT);
            __hip_atomic_store(&bar[1], gen_target, __ATOMIC_RELEASE, __HIP_MEMORY_SCOPE_AGENT);
        } else {
            while (__hip_atomic_load(&bar[1], __ATOMIC_ACQUIRE,
                                     __HIP_MEMORY_SCOPE_AGENT) < gen_target) {
                __builtin_amdgcn_s_sleep(1);
            }
        }
    }
    __syncthreads();
}

// ---------------------------------------------------------------------------
// Phase 0: W[(INSZ+HID) x HID] fp32 row-major -> WxT/WhT bf16 [N][K];
//          h0 -> bf16 ping buffer; reset barrier words.
// ---------------------------------------------------------------------------
__global__ __launch_bounds__(256) void rnn_prep_kernel(
    const float* __restrict__ W, const float* __restrict__ h0,
    bf16_t* __restrict__ WxT, bf16_t* __restrict__ WhT,
    bf16_t* __restrict__ hbuf0, unsigned* __restrict__ bar)
{
    int gid = blockIdx.x * 256 + threadIdx.x;           // 0 .. 2048*1024-1
    int k = gid >> 10;
    int n = gid & (HID - 1);
    bf16_t v = to_bf16(W[gid]);
    if (k < INSZ) WxT[(size_t)n * INSZ + k] = v;
    else          WhT[(size_t)n * HID + (k - INSZ)] = v;
    if (gid < BATCH * HID) hbuf0[gid] = to_bf16(h0[gid]);
    if (gid == 0) { bar[0] = 0u; bar[1] = 0u; }
}

// ---------------------------------------------------------------------------
// Phase 1: x_proj[M=32768][1024] = input @ Wx + b.
// One wave computes a 16(M) x 64(N) tile.  Fragments are double-buffered:
// loads for k-step kk+1 are issued before the WMMAs consuming kk.
// ---------------------------------------------------------------------------
__global__ __launch_bounds__(256) void rnn_xproj_gemm_kernel(
    const float* __restrict__ inp, const bf16_t* __restrict__ WxT,
    const float* __restrict__ bias, float* __restrict__ xproj)
{
    const int lane = threadIdx.x & 31;
    const int w    = blockIdx.x * 8 + (threadIdx.x >> 5);   // 0 .. 32767
    const int n64  = w & 15;          // 64-wide N tile
    const int mt   = w >> 4;          // 16-wide M tile, 0..2047
    const int lm   = lane & 15;
    const int lg   = lane >> 4;

    const float* arow = inp + (size_t)(mt * 16 + lm) * INSZ;
    const bf16_t* b0r = WxT + (size_t)(n64 * 64 +  0 + lm) * INSZ;
    const bf16_t* b1r = WxT + (size_t)(n64 * 64 + 16 + lm) * INSZ;
    const bf16_t* b2r = WxT + (size_t)(n64 * 64 + 32 + lm) * INSZ;
    const bf16_t* b3r = WxT + (size_t)(n64 * 64 + 48 + lm) * INSZ;

    v8f acc0 = {}, acc1 = {}, acc2 = {}, acc3 = {};

    // prologue: fragments for kk = 0
    int kb = 8 * lg;
    v16bf a  = load_frag_f32(arow, kb);
    v16bf f0 = load_frag_bf16(b0r, kb);
    v16bf f1 = load_frag_bf16(b1r, kb);
    v16bf f2 = load_frag_bf16(b2r, kb);
    v16bf f3 = load_frag_bf16(b3r, kb);

    for (int kk = 0; kk < INSZ / 32; ++kk) {
        // issue next iteration's loads first (wraps to kk=0 on last iter;
        // harmless cache-hit reloads keep the loop shape uniform)
        const int kn = ((kk + 1) & 31) * 32 + 8 * lg;
        v16bf an  = load_frag_f32(arow, kn);
        v16bf g0  = load_frag_bf16(b0r, kn);
        v16bf g1  = load_frag_bf16(b1r, kn);
        v16bf g2  = load_frag_bf16(b2r, kn);
        v16bf g3  = load_frag_bf16(b3r, kn);

        acc0 = wmma_bf16(a, f0, acc0);
        acc1 = wmma_bf16(a, f1, acc1);
        acc2 = wmma_bf16(a, f2, acc2);
        acc3 = wmma_bf16(a, f3, acc3);

        a = an; f0 = g0; f1 = g1; f2 = g2; f3 = g3;
    }

    // C/D layout: lane holds column (n64*64 + nt*16 + lm), VGPR r is row
    // (mt*16 + r + 8*lg).  Add bias and store fp32.
    v8f accs[4] = {acc0, acc1, acc2, acc3};
#pragma unroll
    for (int nt = 0; nt < 4; ++nt) {
        const int n = n64 * 64 + nt * 16 + lm;
        const float bv = bias[n];
#pragma unroll
        for (int r = 0; r < 8; ++r) {
            const int m = mt * 16 + r + 8 * lg;
            xproj[(size_t)m * HID + n] = accs[nt][r] + bv;
        }
    }
}

// ---------------------------------------------------------------------------
// Phase 2: persistent recurrence.  256 waves; wave -> one 16x16 output tile
// (4 batch-tiles x 64 hidden-tiles).  Block b owns hidden tiles 2b, 2b+1:
//   - its 32 WhT rows (64 KB) are staged into LDS ONCE (constant all steps)
//   - the full h state (128 KB) is staged global->LDS each step via the
//     ASYNCcnt-tracked async-to-LDS path (no VGPR roundtrip)
//   - the 32-WMMA K-chain then reads only ds_load_b128 (LDS latency),
//     which is what matters for 512 serial steps.
// ---------------------------------------------------------------------------
__global__ __launch_bounds__(256) void rnn_recurrent_kernel(
    const float* __restrict__ xproj, const bf16_t* __restrict__ WhT,
    bf16_t* __restrict__ hA, bf16_t* __restrict__ hB,
    float* __restrict__ out, unsigned* __restrict__ bar)
{
    extern __shared__ char smem_raw[];
    bf16_t* lds_wh = (bf16_t*)smem_raw;                               // 64 KB
    bf16_t* lds_h  = (bf16_t*)(smem_raw + SMEM_WH_ELEMS * sizeof(bf16_t)); // 128 KB

    const int tid  = threadIdx.x;
    const int lane = tid & 31;
    const int wave = tid >> 5;
    const int mt   = (blockIdx.x * 8 + wave) & 3;   // batch tile 0..3
    const int ntl  = wave >> 2;                     // local hidden tile 0/1
    const int lm   = lane & 15;
    const int lg   = lane >> 4;
    const int n    = (blockIdx.x * 2 + ntl) * 16 + lm;   // global hidden col

    // one-time: stage this block's 32 WhT rows into LDS (16B chunks)
    {
        const bf16_t* src = WhT + (size_t)(blockIdx.x * 32) * HID;
#pragma unroll
        for (int i = 0; i < (SMEM_WH_ELEMS / 8) / 256; ++i) {  // 16 iters
            const int c = (tid + i * 256) * 8;
            stage_16B(src + c, lds_wh + c);
        }
        stage_wait();
    }

    bf16_t* hbuf[2] = {hA, hB};
    const bf16_t* browL = lds_wh + (size_t)(ntl * 16 + lm) * HID;
    const bf16_t* arowL = lds_h  + (size_t)(mt * 16 + lm) * HID;

    for (int t = 0; t < SEQ; ++t) {
        const bf16_t* __restrict__ hprev = hbuf[t & 1];
        bf16_t* __restrict__       hnext = hbuf[(t + 1) & 1];

        // stage full h state global -> LDS, cooperatively (16B chunks)
#pragma unroll
        for (int i = 0; i < (SMEM_H_ELEMS / 8) / 256; ++i) {   // 32 iters
            const int c = (tid + i * 256) * 8;
            stage_16B(hprev + c, lds_h + c);
        }

        // accumulator from pre-computed input projection (L2), issued while
        // the LDS staging is still in flight
        v8f acc;
#pragma unroll
        for (int r = 0; r < 8; ++r) {
            const int b = mt * 16 + r + 8 * lg;
            acc[r] = xproj[((size_t)b * SEQ + t) * HID + n];
        }

        stage_wait();
        __syncthreads();   // staged h visible to all waves of the block

        // K = 1024 reduction: all operands from LDS
        for (int kk = 0; kk < HID / 32; ++kk) {
            const int kb = kk * 32 + 8 * lg;
            v16bf a = load_frag_bf16(arowL, kb);
            v16bf b = load_frag_bf16(browL, kb);
            acc = wmma_bf16(a, b, acc);
        }

        // tanh + stores (output sequence, next hidden state, final h_n)
#pragma unroll
        for (int r = 0; r < 8; ++r) {
            const int b = mt * 16 + r + 8 * lg;
            const float v = tanhf(acc[r]);
            out[((size_t)b * SEQ + t) * HID + n] = v;
            hnext[b * HID + n] = to_bf16(v);
            if (t == SEQ - 1)
                out[(size_t)BATCH * SEQ * HID + (size_t)b * HID + n] = v;
        }

        grid_barrier(bar, (unsigned)(t + 1));
    }
}

// ---------------------------------------------------------------------------
extern "C" void kernel_launch(void* const* d_in, const int* in_sizes, int n_in,
                              void* d_out, int out_size, void* d_ws, size_t ws_size,
                              hipStream_t stream)
{
    const float* inp = (const float*)d_in[0];   // (64, 512, 1024) f32
    const float* h0  = (const float*)d_in[1];   // (1, 64, 1024)   f32
    const float* W   = (const float*)d_in[2];   // (2048, 1024)    f32
    const float* b   = (const float*)d_in[3];   // (1024,)         f32
    float* out = (float*)d_out;

    // workspace layout
    char* ws = (char*)d_ws;
    constexpr size_t XPROJ_B = (size_t)BATCH * SEQ * HID * sizeof(float);   // 134 MB
    constexpr size_t WXT_B   = (size_t)INSZ * HID * sizeof(bf16_t);         //   2 MB
    constexpr size_t WHT_B   = (size_t)HID * HID * sizeof(bf16_t);          //   2 MB
    constexpr size_t HBUF_B  = (size_t)BATCH * HID * sizeof(bf16_t);        // 128 KB
    float*    xproj = (float*)(ws);
    bf16_t*   WxT   = (bf16_t*)(ws + XPROJ_B);
    bf16_t*   WhT   = (bf16_t*)(ws + XPROJ_B + WXT_B);
    bf16_t*   hA    = (bf16_t*)(ws + XPROJ_B + WXT_B + WHT_B);
    bf16_t*   hB    = (bf16_t*)(ws + XPROJ_B + WXT_B + WHT_B + HBUF_B);
    unsigned* bar   = (unsigned*)(ws + XPROJ_B + WXT_B + WHT_B + 2 * HBUF_B);

    // Phase 0: weight transpose + bf16 convert, h0 convert, barrier reset
    rnn_prep_kernel<<<dim3((INSZ + HID) * HID / 256), dim3(256), 0, stream>>>(
        W, h0, WxT, WhT, hA, bar);

    // Phase 1: x_proj GEMM (32768 waves, 16x64 tile per wave, pipelined)
    rnn_xproj_gemm_kernel<<<dim3((BATCH * SEQ / 16) * (HID / 64) / 8),
                            dim3(256), 0, stream>>>(inp, WxT, b, xproj);

    // Phase 2: persistent recurrence, 192 KB dynamic LDS per block
    rnn_recurrent_kernel<<<dim3(NBLK_REC), dim3(256), SMEM_BYTES, stream>>>(
        xproj, WhT, hA, hB, out, bar);
}
